// TransformerEncoderLayer_83691732730624
// MI455X (gfx1250) — matrix operand order repaired and checked by reading, not compiled, not run on hardware
//
#include <hip/hip_runtime.h>
#include <hip/hip_bf16.h>
#include <math.h>

typedef _Float16 v16h __attribute__((ext_vector_type(16)));
typedef float    v8f  __attribute__((ext_vector_type(8)));
typedef unsigned v4u  __attribute__((ext_vector_type(4)));
typedef int      v4i  __attribute__((ext_vector_type(4)));
typedef int      v8i  __attribute__((ext_vector_type(8)));

union FragH { v16h v; _Float16 h[16]; unsigned u[8]; uint4 q[2]; };
union FragF { v8f  v; float f[8]; };

#define HEADS 16
#define HD 64

// --------------------------- CDNA5 feature probes ---------------------------
#if defined(__has_builtin)
#if __has_builtin(__builtin_amdgcn_tensor_load_to_lds)
#define HAVE_TDM 1
#endif
#if __has_builtin(__builtin_amdgcn_global_load_async_to_lds_b128)
#define HAVE_ASYNC 1
#endif
#if __has_builtin(__builtin_amdgcn_s_wait_tensorcnt)
#define WAIT_TENSOR0() __builtin_amdgcn_s_wait_tensorcnt(0)
#else
#define WAIT_TENSOR0() asm volatile("s_wait_tensorcnt 0x0" ::: "memory")
#endif
#if __has_builtin(__builtin_amdgcn_s_wait_asynccnt)
#define WAIT_ASYNC0() __builtin_amdgcn_s_wait_asynccnt(0)
#else
#define WAIT_ASYNC0() asm volatile("s_wait_asynccnt 0x0" ::: "memory")
#endif
#endif

#if defined(HAVE_TDM)
#define STAGE_WAIT() WAIT_TENSOR0()
#elif defined(HAVE_ASYNC)
#define STAGE_WAIT() WAIT_ASYNC0()
#else
#define STAGE_WAIT() ((void)0)
#endif

#define AS1_PTR(p) ((__attribute__((address_space(1))) void*)(const void*)(p))
#define AS3_PTR(p) ((__attribute__((address_space(3))) void*)(p))
#define AS3_OFF(p) ((unsigned)(size_t)(__attribute__((address_space(3))) char*)(p))

// ---------------------------------------------------------------------------
// WMMA helpers (CDNA5 ISA layouts, wave32)
// ---------------------------------------------------------------------------
__device__ inline v8f wmma_f16(v16h a, v16h b, v8f c) {
  return __builtin_amdgcn_wmma_f32_16x16x32_f16(false, a, false, b, (short)0, c,
                                                false, false);
}

// A fragment 16x32 f16: lane m=L&15, half g=L>>4; VGPR p holds K pair per ISA.
__device__ inline v16h load_frag_a(const _Float16* base, int ld, int lane) {
  FragH f;
  const _Float16* row = base + (size_t)(lane & 15) * ld;
  int g8 = (lane >> 4) * 8;
#pragma unroll
  for (int p = 0; p < 8; ++p) {
    int kb = ((p < 4) ? (2 * p) : (8 + 2 * p)) + g8;
    f.u[p] = *reinterpret_cast<const unsigned*>(row + kb);
  }
  return f.v;
}

// B fragment 32x16 f16: lane n=L&15, half g=L>>4; K order same as A fragment.
__device__ inline v16h load_frag_b(const _Float16* base, int ld, int lane) {
  FragH f;
  const _Float16* col = base + (lane & 15);
  int g8 = (lane >> 4) * 8;
#pragma unroll
  for (int s = 0; s < 16; ++s) {
    int k = ((s < 8) ? s : (8 + s)) + g8;
    f.h[s] = col[(size_t)k * ld];
  }
  return f.v;
}

__device__ inline v8f zero8() {
  v8f z = {0.f, 0.f, 0.f, 0.f, 0.f, 0.f, 0.f, 0.f};
  return z;
}

// ---------------------------------------------------------------------------
// TDM: DMA a rows x cols (f16) 2-D tile, row stride = stride halves, into LDS.
// D# per CDNA5 ISA 8.3/8.4.  6-arg builtin: (u32x4, i32x8, i32x4, i32x4,
// i32x8, i32 cpol); groups 2/3 + trailing group zero for a 2-D tile.
// ---------------------------------------------------------------------------
#if defined(HAVE_TDM)
__device__ inline void tdm_load_2d(const _Float16* gsrc, unsigned lds_off,
                                   int rows, int cols, int stride_halves) {
  unsigned long long ga = (unsigned long long)(size_t)gsrc;
  v4u g0;
  g0[0] = 1u;                                   // count=1, user descriptor
  g0[1] = lds_off;                              // lds_addr (bytes)
  g0[2] = (unsigned)(ga & 0xffffffffu);         // global_addr[31:0]
  g0[3] = (unsigned)((ga >> 32) & 0x01ffffffu)  // global_addr[56:32]
          | (2u << 30);                         // type = 2 ("image")
  v8i g1;
  g1[0] = 0x00010000;                 // wg_mask=0, data_size=1 (2 bytes)
  g1[1] = 0;                          // atomic_barrier_addr=0, tdim0[15:0]=0
  g1[2] = 0x00004000;                 // tensor_dim0 = 1<<30 (hi), tdim1 lo=0
  g1[3] = 0x00004000 | (cols << 16);  // tensor_dim1 = 1<<30 (hi), tile_dim0
  g1[4] = rows & 0xffff;              // tile_dim1, tile_dim2 = 0
  g1[5] = stride_halves;              // tensor_dim0_stride[31:0]
  g1[6] = 0;                          // stride0[47:32]=0, stride1 lo=0
  g1[7] = 0;
  v4i z4 = {0, 0, 0, 0};
  v8i z8 = {0, 0, 0, 0, 0, 0, 0, 0};
  __builtin_amdgcn_tensor_load_to_lds(g0, g1, z4, z4, z8, 0);
}
#endif

// ---------------------------------------------------------------------------
// wave32 reductions
// ---------------------------------------------------------------------------
__device__ inline float wave_sum(float v) {
#pragma unroll
  for (int o = 16; o > 0; o >>= 1) v += __shfl_xor(v, o, 32);
  return v;
}
__device__ inline float wave_max(float v) {
#pragma unroll
  for (int o = 16; o > 0; o >>= 1) v = fmaxf(v, __shfl_xor(v, o, 32));
  return v;
}

// ---------------------------------------------------------------------------
// fp32 -> fp16 convert
// ---------------------------------------------------------------------------
__global__ void cvt_f16_kernel(const float* __restrict__ src,
                               _Float16* __restrict__ dst, size_t n) {
  size_t i = (size_t)blockIdx.x * blockDim.x + threadIdx.x;
  size_t stride = (size_t)gridDim.x * blockDim.x;
  for (; i < n; i += stride) dst[i] = (_Float16)src[i];
}

// ---------------------------------------------------------------------------
// LayerNorm (ddof=1, eps added to sigma). One block (256 thr) per row, D=1024.
// ---------------------------------------------------------------------------
__global__ void ln_kernel(const float* __restrict__ x, const float* __restrict__ ga,
                          const float* __restrict__ gb, _Float16* __restrict__ out,
                          int D) {
  int row = blockIdx.x, tid = threadIdx.x;
  int lane = tid & 31, w = tid >> 5;
  const float* src = x + (size_t)row * D;
  float v[4], s = 0.f, s2 = 0.f;
#pragma unroll
  for (int i = 0; i < 4; ++i) {
    v[i] = src[tid + i * 256];
    s += v[i];
    s2 += v[i] * v[i];
  }
  __shared__ float ps[8], ps2[8];
  s = wave_sum(s);
  s2 = wave_sum(s2);
  if (lane == 0) { ps[w] = s; ps2[w] = s2; }
  __syncthreads();
  float S1 = 0.f, S2 = 0.f;
#pragma unroll
  for (int k = 0; k < 8; ++k) { S1 += ps[k]; S2 += ps2[k]; }
  float mu = S1 / (float)D;
  float var = fmaxf((S2 - mu * S1) / (float)(D - 1), 0.f);
  float inv = 1.f / (sqrtf(var) + 1e-6f);
#pragma unroll
  for (int i = 0; i < 4; ++i) {
    int j = tid + i * 256;
    out[(size_t)row * D + j] = (_Float16)((v[i] - mu) * inv * ga[j] + gb[j]);
  }
}

// ---------------------------------------------------------------------------
// Generic GEMM: C[M,N] = A[M,K](f16) * W[N,K](f16)^T + bias, 128x128 tile,
// 8 waves/block, each wave -> 32x64 (2x4 WMMA subtiles).
// Double-buffered pipeline: TDM (TENSORcnt) > async b128 (ASYNCcnt) > manual.
// The DMA for tile i+1 is issued before the WMMA burst for tile i, so the
// TDM engine runs concurrently with the matrix pipes.
// MODE 0: f16 out head-major [h][m][d]; 1: f16 [h][d][m]; 2: f32 + residual;
// MODE 3: f16 ReLU.
// ---------------------------------------------------------------------------
template <int MODE>
__global__ void gemm_aw_kernel(const _Float16* __restrict__ A,
                               const _Float16* __restrict__ W,
                               const float* __restrict__ bias,
                               const float* __restrict__ res,
                               void* __restrict__ outv, int M, int N, int K) {
  __shared__ __align__(16) _Float16 As[2][128 * 32];  // A tiles, row-major
  __shared__ __align__(16) _Float16 Ws[2][128 * 32];  // W tiles, row-major
  __shared__ __align__(16) _Float16 Bs[32 * 136];     // transposed [k][n]
  int tid = threadIdx.x;
  int lane = tid & 31, wv = tid >> 5;
  int bm = blockIdx.y * 128, bn = blockIdx.x * 128;
  int wm = (wv >> 1) * 32;  // 0,32,64,96
  int wn = (wv & 1) * 64;   // 0,64

  v8f acc[2][4];
#pragma unroll
  for (int i = 0; i < 2; ++i)
#pragma unroll
    for (int c = 0; c < 4; ++c) acc[i][c] = zero8();

  // Stage one 128x32 A tile + 128x32 W tile into LDS buffer b.
  auto stage = [&](int b, int k0) {
#if defined(HAVE_TDM)
    // Each wave DMAs its own 16-row slice of both tiles (TENSORcnt).
    unsigned asb = AS3_OFF(&As[b][0]) + (unsigned)(wv * 16 * 32 * 2);
    unsigned wsb = AS3_OFF(&Ws[b][0]) + (unsigned)(wv * 16 * 32 * 2);
    tdm_load_2d(A + (size_t)(bm + wv * 16) * K + k0, asb, 16, 32, K);
    tdm_load_2d(W + (size_t)(bn + wv * 16) * K + k0, wsb, 16, 32, K);
#elif defined(HAVE_ASYNC)
    // Per-lane 16B async copies (ASYNCcnt), no VGPR staging.
#pragma unroll
    for (int i = 0; i < 2; ++i) {
      int c = tid + i * 256;
      int r = c >> 2, seg = c & 3;
      __builtin_amdgcn_global_load_async_to_lds_b128(
          AS1_PTR(A + (size_t)(bm + r) * K + k0 + seg * 8),
          AS3_PTR(&As[b][r * 32 + seg * 8]), 0, 0);
      int n = c & 127, sg = c >> 7;
      __builtin_amdgcn_global_load_async_to_lds_b128(
          AS1_PTR(W + (size_t)(bn + n) * K + k0 + sg * 8),
          AS3_PTR(&Ws[b][n * 32 + sg * 8]), 0, 0);
    }
#else
    // Manual fallback: global -> VGPR -> LDS, plus L2 prefetch hint.
#pragma unroll
    for (int i = 0; i < 2; ++i) {
      int c = tid + i * 256;
      int r = c >> 2, seg = c & 3;
      *reinterpret_cast<uint4*>(&As[b][r * 32 + seg * 8]) =
          *reinterpret_cast<const uint4*>(A + (size_t)(bm + r) * K + k0 + seg * 8);
      int n = c & 127, sg = c >> 7;
      *reinterpret_cast<uint4*>(&Ws[b][n * 32 + sg * 8]) =
          *reinterpret_cast<const uint4*>(W + (size_t)(bn + n) * K + k0 + sg * 8);
    }
    if (k0 + 32 < K)
      __builtin_prefetch(A + (size_t)(bm + (tid >> 1)) * K + k0 + 32, 0, 1);
#endif
  };

  stage(0, 0);  // prologue: tile 0 in flight
  int nk = K >> 5;
  for (int ik = 0; ik < nk; ++ik) {
    int b = ik & 1;
    STAGE_WAIT();     // this wave's DMA for buffer b complete
    __syncthreads();  // all waves' slices of buffer b visible; prior reads done

    // LDS->LDS transpose Ws[b][n][k] -> Bs[k][n] for B-fragment reads.
#pragma unroll
    for (int i = 0; i < 2; ++i) {
      int c = tid + i * 256;
      int n = c & 127, sg = c >> 7;
      union { uint4 u; _Float16 h[8]; } t;
      t.u = *reinterpret_cast<const uint4*>(&Ws[b][n * 32 + sg * 8]);
#pragma unroll
      for (int e = 0; e < 8; ++e) Bs[(sg * 8 + e) * 136 + n] = t.h[e];
    }
    if (ik + 1 < nk) stage(b ^ 1, (ik + 1) << 5);  // overlap DMA with WMMA
    __syncthreads();  // Bs visible

    v16h a0 = load_frag_a(&As[b][(wm + 0) * 32], 32, lane);
    v16h a1 = load_frag_a(&As[b][(wm + 16) * 32], 32, lane);
#pragma unroll
    for (int c = 0; c < 4; ++c) {
      v16h bf = load_frag_b(&Bs[wn + c * 16], 136, lane);
      acc[0][c] = wmma_f16(a0, bf, acc[0][c]);
      acc[1][c] = wmma_f16(a1, bf, acc[1][c]);
    }
  }

  int rofs = (lane >> 4) * 8;  // C layout: VGPR r -> M=r (lanes<16) / r+8
  int ncl = lane & 15;
#pragma unroll
  for (int i = 0; i < 2; ++i)
#pragma unroll
    for (int c = 0; c < 4; ++c) {
      FragF fr;
      fr.v = acc[i][c];
#pragma unroll
      for (int r = 0; r < 8; ++r) {
        int mrow = bm + wm + i * 16 + r + rofs;
        int ncol = bn + wn + c * 16 + ncl;
        float val = fr.f[r] + bias[ncol];
        if (MODE == 0) {
          ((_Float16*)outv)[((size_t)(ncol >> 6) * M + mrow) * HD + (ncol & 63)] =
              (_Float16)val;
        } else if (MODE == 1) {
          ((_Float16*)outv)[((size_t)(ncol >> 6) * HD + (ncol & 63)) * M + mrow] =
              (_Float16)val;
        } else if (MODE == 2) {
          ((float*)outv)[(size_t)mrow * N + ncol] =
              val + res[(size_t)mrow * N + ncol];
        } else {
          ((_Float16*)outv)[(size_t)mrow * N + ncol] = (_Float16)fmaxf(val, 0.f);
        }
      }
    }
}

// ---------------------------------------------------------------------------
// Attention scores: Sc[h][q][k] = 0.125 * Q[h][q][:] . K[h][k][:]
// Q head-major [h][S][64]; Kt d-major [h][64][S]. One wave -> 16q x 64k tile.
// ---------------------------------------------------------------------------
__global__ void scores_kernel(const _Float16* __restrict__ Qh,
                              const _Float16* __restrict__ Kt,
                              float* __restrict__ Sc, int S) {
  int KT = S / 64, QT = S / 16;
  int wid = ((int)blockIdx.x * (int)blockDim.x + (int)threadIdx.x) >> 5;
  int lane = threadIdx.x & 31;
  int kt = wid % KT;
  int qt = (wid / KT) % QT;
  int h = wid / (KT * QT);

  const _Float16* Qb = Qh + ((size_t)h * S + (size_t)qt * 16) * HD;
  const _Float16* Kb = Kt + (size_t)h * HD * S + (size_t)kt * 64;

  v8f acc[4];
#pragma unroll
  for (int c = 0; c < 4; ++c) acc[c] = zero8();

#pragma unroll
  for (int ks = 0; ks < 2; ++ks) {  // head dim 64 = 2 k-steps of 32
    v16h a = load_frag_a(Qb + ks * 32, HD, lane);
#pragma unroll
    for (int c = 0; c < 4; ++c) {
      v16h b = load_frag_b(Kb + (size_t)(ks * 32) * S + c * 16, S, lane);
      acc[c] = wmma_f16(a, b, acc[c]);
    }
  }
  int rofs = (lane >> 4) * 8, ncl = lane & 15;
#pragma unroll
  for (int c = 0; c < 4; ++c) {
    FragF fr;
    fr.v = acc[c];
#pragma unroll
    for (int r = 0; r < 8; ++r) {
      int q = qt * 16 + r + rofs;
      int k = kt * 64 + c * 16 + ncl;
      Sc[((size_t)h * S + q) * S + k] = 0.125f * fr.f[r];
    }
  }
}

// ---------------------------------------------------------------------------
// Exact 1.5-entmax per row via tau-bisection (f(tau)=sum relu(x-tau)^2,
// monotone; bracket [-1,0] after x/2 - max shift). Applies key mask.
// One block (256 thr) per (h,q) row, S=2048 -> 8 elems/thread.
// ---------------------------------------------------------------------------
__global__ void entmax_kernel(const float* __restrict__ Sc,
                              const int* __restrict__ mask,
                              _Float16* __restrict__ Ph, int S) {
  int row = blockIdx.x;  // h*S + q
  int tid = threadIdx.x, lane = tid & 31, w = tid >> 5;
  const float* src = Sc + (size_t)row * S;
  _Float16* dst = Ph + (size_t)row * S;
  __shared__ float part[8];

  float x[8];
  float mloc = -3.4e38f;
#pragma unroll
  for (int i = 0; i < 8; ++i) {
    int j = tid + i * 256;
    float s = src[j];
    s = (mask[j] == 0) ? -1e9f : s;
    x[i] = 0.5f * s;
    mloc = fmaxf(mloc, x[i]);
  }
  mloc = wave_max(mloc);
  if (lane == 0) part[w] = mloc;
  __syncthreads();
  float mx = -3.4e38f;
#pragma unroll
  for (int k = 0; k < 8; ++k) mx = fmaxf(mx, part[k]);
  __syncthreads();
#pragma unroll
  for (int i = 0; i < 8; ++i) x[i] -= mx;

  float lo = -1.0f, hi = 0.0f;
  for (int it = 0; it < 32; ++it) {
    float mid = 0.5f * (lo + hi);
    float loc = 0.f;
#pragma unroll
    for (int i = 0; i < 8; ++i) {
      float t = x[i] - mid;
      loc += (t > 0.f) ? t * t : 0.f;
    }
    loc = wave_sum(loc);
    if (lane == 0) part[w] = loc;
    __syncthreads();
    float tot = 0.f;
#pragma unroll
    for (int k = 0; k < 8; ++k) tot += part[k];
    __syncthreads();
    if (tot >= 1.f) lo = mid; else hi = mid;
  }
  float tau = 0.5f * (lo + hi);
#pragma unroll
  for (int i = 0; i < 8; ++i) {
    float t = x[i] - tau;
    t = (t > 0.f) ? t : 0.f;
    dst[tid + i * 256] = (_Float16)(t * t);
  }
}

// ---------------------------------------------------------------------------
// attended = P @ V per head; output merged-head f16 [S][D].
// One wave -> 16q x 64d tile, K=S keys.
// ---------------------------------------------------------------------------
__global__ void pv_kernel(const _Float16* __restrict__ Ph,
                          const _Float16* __restrict__ Vh,
                          _Float16* __restrict__ attnh, int S, int D) {
  int QT = S / 16;
  int wid = ((int)blockIdx.x * (int)blockDim.x + (int)threadIdx.x) >> 5;
  int lane = threadIdx.x & 31;
  int qt = wid % QT;
  int h = wid / QT;

  const _Float16* Pb = Ph + ((size_t)h * S + (size_t)qt * 16) * S;
  const _Float16* Vb = Vh + (size_t)h * S * HD;

  v8f acc[4];
#pragma unroll
  for (int c = 0; c < 4; ++c) acc[c] = zero8();

  for (int k0 = 0; k0 < S; k0 += 32) {
    v16h a = load_frag_a(Pb + k0, S, lane);
#pragma unroll
    for (int c = 0; c < 4; ++c) {
      v16h b = load_frag_b(Vb + (size_t)k0 * HD + c * 16, HD, lane);
      acc[c] = wmma_f16(a, b, acc[c]);
    }
  }
  int rofs = (lane >> 4) * 8, ncl = lane & 15;
#pragma unroll
  for (int c = 0; c < 4; ++c) {
    FragF fr;
    fr.v = acc[c];
#pragma unroll
    for (int r = 0; r < 8; ++r) {
      int q = qt * 16 + r + rofs;
      int d = c * 16 + ncl;
      attnh[(size_t)q * D + h * HD + d] = (_Float16)fr.f[r];
    }
  }
}

// ---------------------------------------------------------------------------
// host-side orchestration
// ---------------------------------------------------------------------------
extern "C" void kernel_launch(void* const* d_in, const int* in_sizes, int n_in,
                              void* d_out, int out_size, void* d_ws, size_t ws_size,
                              hipStream_t stream) {
  const float* inputs = (const float*)d_in[0];
  const int* mask = (const int*)d_in[1];
  const float* wq = (const float*)d_in[2];
  const float* bq = (const float*)d_in[3];
  const float* wk = (const float*)d_in[4];
  const float* bk = (const float*)d_in[5];
  const float* wv = (const float*)d_in[6];
  const float* bv = (const float*)d_in[7];
  const float* wo = (const float*)d_in[8];
  const float* bo = (const float*)d_in[9];
  const float* ln1a = (const float*)d_in[10];
  const float* ln1b = (const float*)d_in[11];
  const float* w1 = (const float*)d_in[12];
  const float* b1 = (const float*)d_in[13];
  const float* w2 = (const float*)d_in[14];
  const float* b2 = (const float*)d_in[15];
  const float* ln2a = (const float*)d_in[16];
  const float* ln2b = (const float*)d_in[17];

  const int S = in_sizes[1];     // 2048
  const int D = in_sizes[3];     // 1024
  const int DFF = in_sizes[13];  // 4096
  const int H = HEADS;

  char* p = (char*)d_ws;
  auto alloc = [&](size_t bytes) -> char* {
    char* r = p;
    p += (bytes + 255) & ~(size_t)255;
    return r;
  };
  _Float16* x1h = (_Float16*)alloc((size_t)S * D * 2);
  _Float16* wqh = (_Float16*)alloc((size_t)D * D * 2);
  _Float16* wkh = (_Float16*)alloc((size_t)D * D * 2);
  _Float16* wvh = (_Float16*)alloc((size_t)D * D * 2);
  _Float16* woh = (_Float16*)alloc((size_t)D * D * 2);
  _Float16* w1h = (_Float16*)alloc((size_t)DFF * D * 2);
  _Float16* w2h = (_Float16*)alloc((size_t)D * DFF * 2);
  _Float16* Qh = (_Float16*)alloc((size_t)S * D * 2);
  _Float16* Kt = (_Float16*)alloc((size_t)S * D * 2);
  _Float16* Vh = (_Float16*)alloc((size_t)S * D * 2);
  float* Sc = (float*)alloc((size_t)H * S * S * 4);
  _Float16* Ph = (_Float16*)alloc((size_t)H * S * S * 2);
  _Float16* attnh = (_Float16*)alloc((size_t)S * D * 2);
  float* y = (float*)alloc((size_t)S * D * 4);
  _Float16* x2h = (_Float16*)alloc((size_t)S * D * 2);
  _Float16* hid = (_Float16*)alloc((size_t)S * DFF * 2);

  cvt_f16_kernel<<<1024, 256, 0, stream>>>(wq, wqh, (size_t)D * D);
  cvt_f16_kernel<<<1024, 256, 0, stream>>>(wk, wkh, (size_t)D * D);
  cvt_f16_kernel<<<1024, 256, 0, stream>>>(wv, wvh, (size_t)D * D);
  cvt_f16_kernel<<<1024, 256, 0, stream>>>(wo, woh, (size_t)D * D);
  cvt_f16_kernel<<<2048, 256, 0, stream>>>(w1, w1h, (size_t)DFF * D);
  cvt_f16_kernel<<<2048, 256, 0, stream>>>(w2, w2h, (size_t)D * DFF);

  ln_kernel<<<S, 256, 0, stream>>>(inputs, ln1a, ln1b, x1h, D);

  dim3 gqkv(D / 128, S / 128);
  gemm_aw_kernel<0><<<gqkv, 256, 0, stream>>>(x1h, wqh, bq, nullptr, Qh, S, D, D);
  gemm_aw_kernel<1><<<gqkv, 256, 0, stream>>>(x1h, wkh, bk, nullptr, Kt, S, D, D);
  gemm_aw_kernel<0><<<gqkv, 256, 0, stream>>>(x1h, wvh, bv, nullptr, Vh, S, D, D);

  int score_waves = H * (S / 16) * (S / 64);
  scores_kernel<<<score_waves / 8, 256, 0, stream>>>(Qh, Kt, Sc, S);

  entmax_kernel<<<H * S, 256, 0, stream>>>(Sc, mask, Ph, S);

  int pv_waves = H * (S / 16);
  pv_kernel<<<pv_waves / 8, 256, 0, stream>>>(Ph, Vh, attnh, S, D);

  gemm_aw_kernel<2><<<gqkv, 256, 0, stream>>>(attnh, woh, bo, inputs, y, S, D, D);

  ln_kernel<<<S, 256, 0, stream>>>(y, ln2a, ln2b, x2h, D);

  dim3 gff1(DFF / 128, S / 128);
  gemm_aw_kernel<3><<<gff1, 256, 0, stream>>>(x2h, w1h, b1, nullptr, hid, S, DFF, D);

  dim3 gff2(D / 128, S / 128);
  gemm_aw_kernel<2><<<gff2, 256, 0, stream>>>(hid, w2h, b2, y, (float*)d_out, S, D,
                                              DFF);
}
